// MLP_15427522527481
// MI455X (gfx1250) — compile-verified, matrix operand order checked by reading, not run on hardware
//
#include <hip/hip_runtime.h>

// B=512, S=2048, D_IN=20, D_H=256, D_OUT=10
// rows = B*S = 1,048,576 ; tiles of 16 rows = 65,536 tiles.

typedef _Float16 v16h __attribute__((ext_vector_type(16)));
typedef float    v8f  __attribute__((ext_vector_type(8)));

#define WAVES_PER_BLOCK 4
#define HPAD 264   // 256 + 8 halves pad -> LDS row stride 528B (4-bank step, conflict-free)

// K index inside a 16x16x32 f16 A/B fragment for element e of v16h, given hi = lane/16.
// From ISA 7.12.2 "16-bit A-Matrix 16x32": lanes0-15 VGPR0..3 -> K0..7, VGPR4..7 -> K16..23;
// lanes16-31: +8.  (B uses the mirrored layout with N = lane%16.)
__device__ __forceinline__ int frag_k(int e, int hi) {
    int v = e >> 1, pos = e & 1;
    return ((v < 4) ? 0 : 16) + hi * 8 + 2 * (v & 3) + pos;
}

// ---------------------------------------------------------------------------
// Kernel 1: zero the global accumulators and pre-pack W1/W2 into WMMA fragment
// order (f16, K padded 20->32 for W1, N padded 10->16 for W2).
//   w1p: [16 ntile][32 lane][16 elem]  halves  (16 KB)
//   w2p: [ 8 kchunk][32 lane][16 elem] halves  ( 8 KB)
// ---------------------------------------------------------------------------
__global__ void mlp_pack_weights(const float* __restrict__ W1,
                                 const float* __restrict__ W2,
                                 _Float16* __restrict__ w1p,
                                 _Float16* __restrict__ w2p,
                                 float* __restrict__ accums) {
    int t = blockIdx.x * blockDim.x + threadIdx.x;
    int nthreads = gridDim.x * blockDim.x;
    if (t == 0) { accums[0] = 0.0f; accums[1] = 0.0f; }

    for (int i = t; i < 16 * 32 * 16; i += nthreads) {        // W1 fragments
        int e = i & 15, lane = (i >> 4) & 31, nt = i >> 9;
        int lo = lane & 15, hi = lane >> 4;
        int K = frag_k(e, hi);                                 // 0..31
        int n = nt * 16 + lo;                                  // 0..255
        float val = (K < 20) ? W1[K * 256 + n] : 0.0f;         // pad K>=20
        w1p[i] = (_Float16)val;
    }
    for (int i = t; i < 8 * 32 * 16; i += nthreads) {          // W2 fragments
        int e = i & 15, lane = (i >> 4) & 31, kc = i >> 9;
        int lo = lane & 15, hi = lane >> 4;
        int K = kc * 32 + frag_k(e, hi);                       // 0..255
        float val = (lo < 10) ? W2[K * 10 + lo] : 0.0f;        // pad out-cols
        w2p[i] = (_Float16)val;
    }
}

// ---------------------------------------------------------------------------
// Kernel 2: fused MLP. One wave = one 16-row tile. GEMM1 (K=32 padded) with
// 16 WMMAs, bias+ReLU, transpose h through per-wave LDS, GEMM2 (K=256) with
// 8 WMMAs, then sum / abs-sum reduction to two global float atomics per block.
// ---------------------------------------------------------------------------
__global__ __launch_bounds__(32 * WAVES_PER_BLOCK)
void mlp_main(const float* __restrict__ X,
              const float* __restrict__ b1,
              const float* __restrict__ b2,
              const _Float16* __restrict__ w1p,
              const _Float16* __restrict__ w2p,
              float* __restrict__ accums) {
    __shared__ _Float16 lds_h[WAVES_PER_BLOCK][16][HPAD];
    __shared__ float blk[2];

    const int lane = threadIdx.x & 31;
    const int wave = threadIdx.x >> 5;
    const int lo = lane & 15, hi = lane >> 4;

    if (threadIdx.x == 0) { blk[0] = 0.0f; blk[1] = 0.0f; }

    const long tile = (long)blockIdx.x * WAVES_PER_BLOCK + wave;
    const long row  = tile * 16 + lo;                 // row in [0, B*S)
    const float* xr = X + row * 20;

    // Hint the next tile's rows toward the caches while we compute.
    __builtin_prefetch(xr + (long)WAVES_PER_BLOCK * 16 * 20, 0, 0);

    // ---- A fragment for this X tile (f32 -> f16, K padded to 32) ----
    v16h a;
#pragma unroll
    for (int e = 0; e < 16; ++e) {
        int K = frag_k(e, hi);
        a[e] = (K < 20) ? (_Float16)xr[K] : (_Float16)0.0f;
    }

    // ---- GEMM1: h[16x256] = relu(X*W1 + b1), one WMMA per 16-col tile ----
#pragma unroll 1
    for (int nt = 0; nt < 16; ++nt) {
        v16h bfrag = *(const v16h*)(w1p + (nt * 32 + lane) * 16);  // 32B aligned
        v8f c = {};
        c = __builtin_amdgcn_wmma_f32_16x16x32_f16(
                false, a, false, bfrag, (short)0, c, false, false);
        float bias = b1[nt * 16 + lo];
#pragma unroll
        for (int vv = 0; vv < 8; ++vv) {
            float hval = fmaxf(c[vv] + bias, 0.0f);
            // C/D layout: this lane owns h[row vv+8*hi][col nt*16+lo]
            lds_h[wave][vv + 8 * hi][nt * 16 + lo] = (_Float16)hval;
        }
    }

    // ---- GEMM2: y[16x16pad] = h * W2pad, K=256 in 8 chunks of 32 ----
    v8f y = {};
#pragma unroll 1
    for (int kc = 0; kc < 8; ++kc) {
        v16h ah;
#pragma unroll
        for (int e = 0; e < 16; ++e) {
            int K = kc * 32 + frag_k(e, hi);
            ah[e] = lds_h[wave][lo][K];               // A layout: row = lo
        }
        v16h bw = *(const v16h*)(w2p + (kc * 32 + lane) * 16);
        y = __builtin_amdgcn_wmma_f32_16x16x32_f16(
                false, ah, false, bw, (short)0, y, false, false);
    }

    // ---- per-lane partial sum / abs-sum over the 10 valid output cols ----
    float s = 0.0f, sa = 0.0f;
    if (lo < 10) {
        float bias2 = b2[lo];
#pragma unroll
        for (int vv = 0; vv < 8; ++vv) {
            float yv = y[vv] + bias2;
            s  += yv;
            sa += fabsf(yv);
        }
    }
#pragma unroll
    for (int off = 16; off > 0; off >>= 1) {
        s  += __shfl_xor(s,  off, 32);
        sa += __shfl_xor(sa, off, 32);
    }
    __syncthreads();
    if (lane == 0) { atomicAdd(&blk[0], s); atomicAdd(&blk[1], sa); }
    __syncthreads();
    if (threadIdx.x == 0) {
        atomicAdd(&accums[0], blk[0]);
        atomicAdd(&accums[1], blk[1]);
    }
}

// ---------------------------------------------------------------------------
// Kernel 3: replicate the reference's halving loop on the abs-sum, then scale.
// ---------------------------------------------------------------------------
__global__ void mlp_finalize(const float* __restrict__ accums,
                             float* __restrict__ out) {
    float s  = accums[0];
    float sa = accums[1];
    int n = 0;
    while (sa > 1.0f && n < 256) { sa *= 0.5f; ++n; }
    out[0] = ldexpf(s, -n);
}

extern "C" void kernel_launch(void* const* d_in, const int* in_sizes, int n_in,
                              void* d_out, int out_size, void* d_ws, size_t ws_size,
                              hipStream_t stream) {
    (void)in_sizes; (void)n_in; (void)out_size; (void)ws_size;
    const float* X  = (const float*)d_in[0];   // [512,2048,20]
    const float* W1 = (const float*)d_in[1];   // [20,256]
    const float* b1 = (const float*)d_in[2];   // [256]
    const float* W2 = (const float*)d_in[3];   // [256,10]
    const float* b2 = (const float*)d_in[4];   // [10]
    float* out = (float*)d_out;

    char* ws = (char*)d_ws;
    float*    accums = (float*)ws;                       // 2 floats (32B slot)
    _Float16* w1p    = (_Float16*)(ws + 32);             // 16 KB
    _Float16* w2p    = (_Float16*)(ws + 32 + 16 * 1024); //  8 KB

    mlp_pack_weights<<<48, 256, 0, stream>>>(W1, W2, w1p, w2p, accums);

    const long tiles = (512L * 2048L) / 16;              // 65,536 tiles
    mlp_main<<<tiles / WAVES_PER_BLOCK, 32 * WAVES_PER_BLOCK, 0, stream>>>(
        X, b1, b2, w1p, w2p, accums);

    mlp_finalize<<<1, 1, 0, stream>>>(accums, out);
}
